// DeltaNet_31877247271530
// MI455X (gfx1250) — matrix operand order repaired and checked by reading, not compiled, not run on hardware
//
#include <hip/hip_runtime.h>
#include <hip/hip_bf16.h>

// DeltaNet chunkwise forward for MI455X (gfx1250, wave32, WMMA).
// Phase 1 (chunk-parallel): per (b,h,chunk): normalize k, A=strict_tril(kb@kn^T),
//   invert unit-lower (I+A), u=inv@(v*beta), w=inv@kb  -> workspace.
// Phase 2 (scan, dv-sliced): per (b,h,dv_slice of 64): carry S (256x64) in LDS,
//   att=tril(qn@kn^T); u_adj=u - w@S; o=qn@S + att@u_adj; S += kn^T@u_adj.
// All GEMMs via v_wmma_f32_16x16x32_bf16 (f32 accumulate, bf16 operands).
// Scan kernel prefetches chunk ci+1's operands (global_prefetch_b8) to hide
// HBM latency behind the current chunk's ~230 WMMAs (only 64 blocks in flight).

typedef __bf16 bf16_t;
typedef bf16_t v16bf __attribute__((ext_vector_type(16)));
typedef float  v8f   __attribute__((ext_vector_type(8)));

#define C    32       // chunk
#define DK   256
#define DV   256
#define L    4096
#define NCH  128      // L / C
#define BH   16       // b*h
#define NSL  4        // dv slices
#define DVS  64       // dv per slice

// ---------------- WMMA fragment helpers (CDNA5 16x16x32 bf16 layouts) --------

// A fragment: 16(M) x 32(K) tile, memory row-major M x K with row stride ld.
// lane 0-15: M=lane, K in {0..7,16..23}; lane 16-31: M=lane-16, K in {8..15,24..31}.
__device__ __forceinline__ v16bf fragA(const bf16_t* p, int ld, int lane) {
  int m = lane & 15, hi = (lane >> 4) & 1;
  v16bf f;
#pragma unroll
  for (int v = 0; v < 8; ++v) {
    int kk = ((v >> 2) << 4) + (hi << 3) + ((v & 3) << 1);
    f[2 * v]     = p[m * ld + kk];
    f[2 * v + 1] = p[m * ld + kk + 1];
  }
  return f;
}

// A fragment from transposed storage: memory holds K x M, element (m,k)=p[k*ld+m].
__device__ __forceinline__ v16bf fragAT(const bf16_t* p, int ld, int lane) {
  int m = lane & 15, hi = (lane >> 4) & 1;
  v16bf f;
#pragma unroll
  for (int v = 0; v < 8; ++v) {
    int kk = ((v >> 2) << 4) + (hi << 3) + ((v & 3) << 1);
    f[2 * v]     = p[kk * ld + m];
    f[2 * v + 1] = p[(kk + 1) * ld + m];
  }
  return f;
}

// B fragment: 32(K) x 16(N) tile, memory row-major K x N with row stride ld.
// lanes 0-15: N=lane, K in {0..7,16..23}; lanes 16-31: N=lane-16, K in {8..15,24..31}; 2 K per VGPR.
__device__ __forceinline__ v16bf fragB(const bf16_t* p, int ld, int lane) {
  int n = lane & 15, base = ((lane >> 4) & 1) << 3;
  v16bf f;
#pragma unroll
  for (int v = 0; v < 8; ++v) {
    int kk = ((v >> 2) << 4) + base + ((v & 3) << 1);
    f[2 * v]     = p[kk * ld + n];
    f[2 * v + 1] = p[(kk + 1) * ld + n];
  }
  return f;
}

// B fragment from transposed storage: memory holds N x K, element (k,n)=p[n*ld+k].
__device__ __forceinline__ v16bf fragBT(const bf16_t* p, int ld, int lane) {
  int n = lane & 15, base = ((lane >> 4) & 1) << 3;
  v16bf f;
#pragma unroll
  for (int v = 0; v < 8; ++v) {
    int kk = ((v >> 2) << 4) + base + ((v & 3) << 1);
    f[2 * v]     = p[n * ld + kk];
    f[2 * v + 1] = p[n * ld + kk + 1];
  }
  return f;
}

// C/D fragment: lane n=lane&15; vgpr r holds row (r + 8*(lane>=16)).
__device__ __forceinline__ v8f loadC(const float* p, int ld, int lane) {
  int n = lane & 15, hi = (lane >> 4) & 1;
  v8f c;
#pragma unroll
  for (int r = 0; r < 8; ++r) c[r] = p[(r + 8 * hi) * ld + n];
  return c;
}
__device__ __forceinline__ void storeC(float* p, int ld, int lane, v8f c) {
  int n = lane & 15, hi = (lane >> 4) & 1;
#pragma unroll
  for (int r = 0; r < 8; ++r) p[(r + 8 * hi) * ld + n] = c[r];
}

__device__ __forceinline__ v8f wmma_bf16(v16bf a, v16bf b, v8f c) {
  return __builtin_amdgcn_wmma_f32_16x16x32_bf16(false, a, false, b, (short)0, c,
                                                 false, false);
}

// Prefetch a [bytes]-sized contiguous region, one 128B line per thread-step.
// Lowers to global_prefetch_b8 (no counters, no VGPR return) on gfx1250.
__device__ __forceinline__ void prefetch_region(const char* p, int bytes, int t) {
  for (int off = t * 128; off < bytes; off += 256 * 128)
    __builtin_prefetch(p + off, 0, 1);
}

// ---------------- Kernel 1: chunk prep ---------------------------------------
// grid = BH*NCH blocks, 256 threads (8 waves).
// Dynamic LDS layout (bytes):
//   stage f32 [0,32768)  kn_bf [32768,49152)  kb_bf [49152,65536)
//   vb_bf [65536,81920)  A f32 [81920,86016)  inv_bf [86016,88064)
//   betas [88064,88192)  norms [88192,88320)
#define LDS1 88320

__global__ __launch_bounds__(256) void deltanet_prep(
    const float* __restrict__ q, const float* __restrict__ k,
    const float* __restrict__ v, const float* __restrict__ beta,
    float* __restrict__ u_ws, bf16_t* __restrict__ w_ws) {
  (void)q;
  extern __shared__ char smem[];
  float*  stage  = (float*)(smem);
  bf16_t* kn_bf  = (bf16_t*)(smem + 32768);
  bf16_t* kb_bf  = (bf16_t*)(smem + 49152);
  bf16_t* vb_bf  = (bf16_t*)(smem + 65536);
  float*  Amat   = (float*)(smem + 81920);
  bf16_t* inv_bf = (bf16_t*)(smem + 86016);
  float*  betas  = (float*)(smem + 88064);
  float*  norms  = (float*)(smem + 88192);

  const int t = threadIdx.x, lane = t & 31, wv = t >> 5;
  const int bh = blockIdx.x / NCH, ci = blockIdx.x % NCH;
  const size_t cbase = ((size_t)bh * NCH + ci) * (size_t)(C * DK);

  if (t < C) betas[t] = beta[(size_t)bh * L + (size_t)ci * C + t];

  // stage k chunk, normalize rows, build kn(bf16) and kb=kn*beta(bf16)
  for (int e = t; e < C * DK; e += 256) stage[e] = k[cbase + e];
  __syncthreads();
  if (t < C) {
    float s = 0.f;
    for (int j = 0; j < DK; ++j) { float x = stage[t * DK + j]; s += x * x; }
    norms[t] = rsqrtf(s);
  }
  __syncthreads();
  for (int e = t; e < C * DK; e += 256) {
    int r = e >> 8;
    float kn = stage[e] * norms[r];
    kn_bf[e] = (bf16_t)kn;
    kb_bf[e] = (bf16_t)(kn * betas[r]);
  }
  __syncthreads();

  // stage v chunk, vb = v*beta
  for (int e = t; e < C * DV; e += 256) stage[e] = v[cbase + e];
  __syncthreads();
  for (int e = t; e < C * DV; e += 256) {
    int r = e >> 8;
    vb_bf[e] = (bf16_t)(stage[e] * betas[r]);
  }
  __syncthreads();

  // A = strict_tril(kb @ kn^T)  (32x32, K=256): 2x2 tiles on waves 0..3
  if (wv < 4) {
    const int tm = wv >> 1, tn = wv & 1;
    v8f acc = {};
#pragma unroll
    for (int kk = 0; kk < 8; ++kk) {
      v16bf a = fragA(kb_bf + tm * 16 * DK + kk * 32, DK, lane);
      v16bf b = fragBT(kn_bf + (tn * 16) * DK + kk * 32, DK, lane);
      acc = wmma_bf16(a, b, acc);
    }
    const int n = lane & 15, hi = (lane >> 4) & 1;
#pragma unroll
    for (int r = 0; r < 8; ++r) {
      int i = tm * 16 + r + 8 * hi, j = tn * 16 + n;
      Amat[i * C + j] = (j < i) ? acc[r] : 0.f;
    }
  }
  __syncthreads();

  // inv = (I + A)^{-1}: unit-lower triangular forward substitution, col per lane
  if (t < C) {
    float X[C];
    const int j = t;
    for (int i = 0; i < C; ++i) {
      float s = (i == j) ? 1.f : 0.f;
      for (int m = j; m < i; ++m) s -= Amat[i * C + m] * X[m];
      X[i] = s;
    }
    for (int i = 0; i < C; ++i) inv_bf[i * C + j] = (bf16_t)X[i];
  }
  __syncthreads();

  // u = inv @ vb (f32 out), w = inv @ kb (bf16 out). 2x16 tiles, 4 per wave.
  float*  u_out = u_ws + cbase;
  bf16_t* w_out = w_ws + cbase;
  const int n = lane & 15, hi = (lane >> 4) & 1;
#pragma unroll
  for (int s = 0; s < 4; ++s) {
    const int tid = wv * 4 + s, tm = tid >> 4, tn = tid & 15;
    v16bf a = fragA(inv_bf + tm * 16 * C, C, lane);  // K=32
    {
      v16bf b = fragB(vb_bf + tn * 16, DV, lane);
      v8f acc = {};
      acc = wmma_bf16(a, b, acc);
#pragma unroll
      for (int r = 0; r < 8; ++r)
        u_out[(size_t)(tm * 16 + r + 8 * hi) * DV + tn * 16 + n] = acc[r];
    }
    {
      v16bf b = fragB(kb_bf + tn * 16, DK, lane);
      v8f acc = {};
      acc = wmma_bf16(a, b, acc);
#pragma unroll
      for (int r = 0; r < 8; ++r)
        w_out[(size_t)(tm * 16 + r + 8 * hi) * DK + tn * 16 + n] = (bf16_t)acc[r];
    }
  }
}

// ---------------- Kernel 2: sequential scan over chunks ----------------------
// grid = BH*NSL blocks, 256 threads. S slice (256 x 64 f32) lives in LDS.
// Dynamic LDS layout (bytes):
//   S f32 [0,65536)   S_bf|stage [65536,98304)   qn_bf [98304,114688)
//   kn_bf [114688,131072)   w_bf [131072,147456)   u_s f32 [147456,155648)
//   uadj_bf [155648,159744) att_bf [159744,161792) norms [161792,161920)
#define LDS2 161920

__global__ __launch_bounds__(256) void deltanet_scan(
    const float* __restrict__ q, const float* __restrict__ k,
    const float* __restrict__ u_ws, const bf16_t* __restrict__ w_ws,
    float* __restrict__ out, float* __restrict__ S_out) {
  extern __shared__ char smem[];
  float*  S       = (float*)(smem);
  bf16_t* S_bf    = (bf16_t*)(smem + 65536);
  float*  stage   = (float*)(smem + 65536);  // aliases S_bf (phase-disjoint)
  bf16_t* qn_bf   = (bf16_t*)(smem + 98304);
  bf16_t* kn_bf   = (bf16_t*)(smem + 114688);
  bf16_t* w_bf    = (bf16_t*)(smem + 131072);
  float*  u_s     = (float*)(smem + 147456);
  bf16_t* uadj_bf = (bf16_t*)(smem + 155648);
  bf16_t* att_bf  = (bf16_t*)(smem + 159744);
  float*  norms   = (float*)(smem + 161792);

  const int t = threadIdx.x, lane = t & 31, wv = t >> 5;
  const int bh = blockIdx.x >> 2, slice = blockIdx.x & (NSL - 1);
  const int col0 = slice * DVS;
  const int n = lane & 15, hi = (lane >> 4) & 1;

  for (int e = t; e < DK * DVS; e += 256) S[e] = 0.f;
  __syncthreads();

  for (int ci = 0; ci < NCH; ++ci) {
    const size_t cbase = ((size_t)bh * NCH + ci) * (size_t)(C * DK);

    // ---- normalize q chunk -> qn_bf (stage aliases S_bf; snapshot comes later)
    for (int e = t; e < C * DK; e += 256) stage[e] = q[cbase + e];
    __syncthreads();
    if (t < C) {
      float s = 0.f;
      for (int j = 0; j < DK; ++j) { float x = stage[t * DK + j]; s += x * x; }
      norms[t] = rsqrtf(s);
    }
    __syncthreads();
    for (int e = t; e < C * DK; e += 256) {
      int r = e >> 8;
      qn_bf[e] = (bf16_t)(stage[e] * norms[r]);
    }
    __syncthreads();
    // ---- normalize k chunk -> kn_bf
    for (int e = t; e < C * DK; e += 256) stage[e] = k[cbase + e];
    __syncthreads();
    if (t < C) {
      float s = 0.f;
      for (int j = 0; j < DK; ++j) { float x = stage[t * DK + j]; s += x * x; }
      norms[t] = rsqrtf(s);
    }
    __syncthreads();
    for (int e = t; e < C * DK; e += 256) {
      int r = e >> 8;
      kn_bf[e] = (bf16_t)(stage[e] * norms[r]);
    }
    __syncthreads();

    // ---- snapshot S -> bf16 (overwrites stage), stage w chunk + u slice
    for (int e = t; e < DK * DVS; e += 256) S_bf[e] = (bf16_t)S[e];
    const bf16_t* wsrc = w_ws + cbase;
    for (int e = t; e < C * DK; e += 256) w_bf[e] = wsrc[e];
    const float* usrc = u_ws + cbase;
    for (int e = t; e < C * DVS; e += 256) {
      int r = e >> 6, cc = e & (DVS - 1);
      u_s[e] = usrc[(size_t)r * DV + col0 + cc];
    }

    // ---- prefetch chunk ci+1 operands; overlaps this chunk's WMMA work.
    if (ci + 1 < NCH) {
      const size_t nbase = cbase + (size_t)(C * DK);
      prefetch_region((const char*)(q + nbase), C * DK * 4, t);
      prefetch_region((const char*)(k + nbase), C * DK * 4, t);
      prefetch_region((const char*)(w_ws + nbase), C * DK * 2, t);
      prefetch_region((const char*)(u_ws + nbase), C * DV * 4, t);
    }
    __syncthreads();

    // ---- att = tril(qn @ kn^T) incl diag: 2x2 tiles on waves 0..3 (K=256)
    if (wv < 4) {
      const int tm = wv >> 1, tn = wv & 1;
      v8f acc = {};
#pragma unroll
      for (int kk = 0; kk < 8; ++kk) {
        v16bf a = fragA(qn_bf + tm * 16 * DK + kk * 32, DK, lane);
        v16bf b = fragBT(kn_bf + (tn * 16) * DK + kk * 32, DK, lane);
        acc = wmma_bf16(a, b, acc);
      }
#pragma unroll
      for (int r = 0; r < 8; ++r) {
        int i = tm * 16 + r + 8 * hi, j = tn * 16 + n;
        att_bf[i * C + j] = (bf16_t)((j <= i) ? acc[r] : 0.f);
      }
    }
    __syncthreads();

    // ---- u_adj = u_s - w @ S : 2x4 tiles, one per wave (K=256)
    {
      const int tm = wv >> 2, tn = wv & 3;
      v8f acc = {};
#pragma unroll
      for (int kk = 0; kk < 8; ++kk) {
        v16bf a = fragA(w_bf + tm * 16 * DK + kk * 32, DK, lane);
        v16bf b = fragB(S_bf + (kk * 32) * DVS + tn * 16, DVS, lane);
        acc = wmma_bf16(a, b, acc);
      }
#pragma unroll
      for (int r = 0; r < 8; ++r) {
        int i = tm * 16 + r + 8 * hi, j = tn * 16 + n;
        uadj_bf[i * DVS + j] = (bf16_t)(u_s[i * DVS + j] - acc[r]);
      }
    }
    __syncthreads();

    // ---- o = qn @ S + att @ u_adj : 2x4 tiles, one per wave
    {
      const int tm = wv >> 2, tn = wv & 3;
      v8f acc = {};
#pragma unroll
      for (int kk = 0; kk < 8; ++kk) {
        v16bf a = fragA(qn_bf + tm * 16 * DK + kk * 32, DK, lane);
        v16bf b = fragB(S_bf + (kk * 32) * DVS + tn * 16, DVS, lane);
        acc = wmma_bf16(a, b, acc);
      }
      {
        v16bf a = fragA(att_bf + tm * 16 * C, C, lane);  // K=32
        v16bf b = fragB(uadj_bf + tn * 16, DVS, lane);
        acc = wmma_bf16(a, b, acc);
      }
      const size_t obase = ((size_t)bh * L + (size_t)ci * C) * (size_t)DV;
#pragma unroll
      for (int r = 0; r < 8; ++r) {
        int i = tm * 16 + r + 8 * hi;
        out[obase + (size_t)i * DV + col0 + tn * 16 + n] = acc[r];
      }
    }

    // ---- S += kn^T @ u_adj : 16x4 tiles, 8 per wave (K=32); wave-disjoint rows
#pragma unroll
    for (int ss = 0; ss < 2; ++ss) {
      const int tm = wv * 2 + ss;
      v16bf a = fragAT(kn_bf + tm * 16, DK, lane);  // (dk x c) from (c x dk)
#pragma unroll
      for (int tn = 0; tn < 4; ++tn) {
        v16bf b = fragB(uadj_bf + tn * 16, DVS, lane);
        float* st = S + (tm * 16) * DVS + tn * 16;
        v8f cacc = loadC(st, DVS, lane);
        cacc = wmma_bf16(a, b, cacc);
        storeC(st, DVS, lane, cacc);
      }
    }
    __syncthreads();  // protects S_bf/stage alias + S snapshot next iter
  }

  // ---- write final state slice: S_out[bh, :, col0:col0+64]
  float* Sg = S_out + (size_t)bh * DK * DV;
  for (int e = t; e < DK * DVS; e += 256) {
    int r = e >> 6, cc = e & (DVS - 1);
    Sg[(size_t)r * DV + col0 + cc] = S[e];
  }
}

// ---------------- launch ------------------------------------------------------
extern "C" void kernel_launch(void* const* d_in, const int* in_sizes, int n_in,
                              void* d_out, int out_size, void* d_ws, size_t ws_size,
                              hipStream_t stream) {
  (void)in_sizes; (void)n_in; (void)out_size; (void)ws_size;
  const float* q    = (const float*)d_in[0];
  const float* k    = (const float*)d_in[1];
  const float* v    = (const float*)d_in[2];
  const float* beta = (const float*)d_in[3];

  float* out   = (float*)d_out;
  float* S_out = out + (size_t)BH * L * DV;  // outputs concatenated: out, S

  // workspace: u (f32, 64 MB) then w (bf16, 32 MB)
  float*  u_ws = (float*)d_ws;
  bf16_t* w_ws = (bf16_t*)((char*)d_ws + (size_t)BH * L * DV * sizeof(float));

  deltanet_prep<<<BH * NCH, 256, LDS1, stream>>>(q, k, v, beta, u_ws, w_ws);
  deltanet_scan<<<BH * NSL, 256, LDS2, stream>>>(q, k, u_ws, w_ws, out, S_out);
}
